// FactorAttnConvRelPosEnc_45930380263593
// MI455X (gfx1250) — matrix operand order repaired and checked by reading, not compiled, not run on hardware
//
#include <hip/hip_runtime.h>
#include <hip/hip_bf16.h>
#include <math.h>

// ---------------------------------------------------------------------------
// FactorAttnConvRelPosEnc for MI455X (gfx1250), bf16 WMMA pipeline
// Round 3: async global->LDS double buffering (fixed builtin pointer types).
// ---------------------------------------------------------------------------

typedef __attribute__((ext_vector_type(16))) __bf16 v16bf;
typedef __attribute__((ext_vector_type(8)))  float  v8f;

#define BDIM 256
#define BB   16
#define NTOK 3137
#define CC   512
#define NH   8
#define CH   64
#define HWP  3136
#define MALL (BB*NTOK)   /* 50192, multiple of 16 */
#define MT197 197        /* ceil(3137/16) */

// Async-to-LDS builtins (gfx1250); fall back to sync staging if absent.
#if defined(__has_builtin)
#if __has_builtin(__builtin_amdgcn_global_load_async_to_lds_b128) && \
    __has_builtin(__builtin_amdgcn_s_wait_asynccnt)
#define HAVE_ASYNC_LDS 1
#endif
#endif

#ifdef HAVE_ASYNC_LDS
// Builtin expects: (v4i addrspace(1)*, v4i addrspace(3)*, imm offset, imm cpol)
typedef int v4i_ __attribute__((vector_size(4 * sizeof(int))));
#define AS1CAST(p) ((__attribute__((address_space(1))) v4i_*)(p))
#define AS3CAST(p) ((__attribute__((address_space(3))) v4i_*)(p))
#endif

__device__ __forceinline__ unsigned short f2bf(float f){
  unsigned int u = __float_as_uint(f);
  u += 0x7fffu + ((u >> 16) & 1u);         // round-to-nearest-even
  return (unsigned short)(u >> 16);
}
__device__ __forceinline__ float bf2f(unsigned short u){
  return __uint_as_float(((unsigned int)u) << 16);
}

union Frag { v16bf v; unsigned short u[16]; };
union Acc  { v8f   v; float f[8]; };

__device__ __forceinline__ void unpack8(uint4 p, unsigned short* d){
  d[0]=(unsigned short)(p.x & 0xffff); d[1]=(unsigned short)(p.x >> 16);
  d[2]=(unsigned short)(p.y & 0xffff); d[3]=(unsigned short)(p.y >> 16);
  d[4]=(unsigned short)(p.z & 0xffff); d[5]=(unsigned short)(p.z >> 16);
  d[6]=(unsigned short)(p.w & 0xffff); d[7]=(unsigned short)(p.w >> 16);
}

// ---------------------------------------------------------------------------
// f32 -> bf16 conversion, 8 elements/thread (all sizes are multiples of 8)
// ---------------------------------------------------------------------------
__global__ void cvt_f32_bf16(const float* __restrict__ src,
                             unsigned short* __restrict__ dst, long long n8){
  long long i = (long long)blockIdx.x * blockDim.x + threadIdx.x;
  long long stride = (long long)gridDim.x * blockDim.x;
  for (; i < n8; i += stride) {
    const float4 a = ((const float4*)src)[i * 2];
    const float4 b = ((const float4*)src)[i * 2 + 1];
    uint4 o;
    o.x = (unsigned)f2bf(a.x) | ((unsigned)f2bf(a.y) << 16);
    o.y = (unsigned)f2bf(a.z) | ((unsigned)f2bf(a.w) << 16);
    o.z = (unsigned)f2bf(b.x) | ((unsigned)f2bf(b.y) << 16);
    o.w = (unsigned)f2bf(b.z) | ((unsigned)f2bf(b.w) << 16);
    ((uint4*)dst)[i] = o;
  }
}

// ---------------------------------------------------------------------------
// Per-wave 16x16x32 tile compute from staged LDS tiles (inlined -> ds ops)
// ---------------------------------------------------------------------------
__device__ __forceinline__ void wmma_tiles(const unsigned short (*Asb)[32],
                                           const unsigned short (*Bsb)[32],
                                           int lane, int wm, int wn, Acc* acc){
  // A fragment: lanes 0-15 hold M=lane, K in {0..7, 16..23}; lanes 16-31 +8
  Frag af;
  const int ml = wm * 16 + (lane & 15);
  const int k0 = (lane < 16) ? 0 : 8;
  unpack8(*(const uint4*)(&Asb[ml][k0]),      &af.u[0]);
  unpack8(*(const uint4*)(&Asb[ml][k0 + 16]), &af.u[8]);
  #pragma unroll
  for (int t = 0; t < 2; ++t) {
    // B fragment: lanes 0-15 hold K in {0..7, 16..23}; lanes 16-31 +8
    Frag bfr;
    const int nl  = wn * 32 + t * 16 + (lane & 15);
    const int kb0 = (lane < 16) ? 0 : 8;
    unpack8(*(const uint4*)(&Bsb[nl][kb0]),     &bfr.u[0]);
    unpack8(*(const uint4*)(&Bsb[nl][kb0 + 16]), &bfr.u[8]);
    acc[t].v = __builtin_amdgcn_wmma_f32_16x16x32_bf16(
        false, af.v, false, bfr.v, (short)0, acc[t].v, false, false);
  }
}

// ---------------------------------------------------------------------------
// Big GEMM:  C[M,Nn] = A[M,K] @ W[Nn,K]^T + bias   (A,W bf16; acc f32)
// mode 0: qkv epilogue -> scatter bf16 into q/k/v (B,h,N,Ch) layout
// mode 1: proj epilogue -> f32 out[M,Nn]
// Block tile 64x64, K-step 32, 8 waves; double-buffered async global->LDS.
// ---------------------------------------------------------------------------
__global__ void gemm_bf16_tn(const unsigned short* __restrict__ A,
                             const unsigned short* __restrict__ W,
                             const float* __restrict__ bias,
                             int M, int Nn, int K, int mode,
                             unsigned short* __restrict__ qb,
                             unsigned short* __restrict__ kbuf,
                             unsigned short* __restrict__ vbuf,
                             float* __restrict__ outf)
{
  __shared__ __align__(16) unsigned short As[2][64][32];
  __shared__ __align__(16) unsigned short Bs[2][64][32];

  const int tid  = threadIdx.x;
  const int lane = tid & 31;
  const int w    = tid >> 5;
  const int wm   = w >> 1;        // 0..3  (M sub-tile)
  const int wn   = w & 1;         // 0..1  (N sub-tile pair)
  const int m0   = blockIdx.y * 64;
  const int n0   = blockIdx.x * 64;

  Acc acc[2];
  #pragma unroll
  for (int t = 0; t < 2; ++t)
    #pragma unroll
    for (int i = 0; i < 8; ++i) acc[t].f[i] = 0.0f;

  const int lr = tid >> 2;        // 0..63 staging row
  const int lc = (tid & 3) * 8;   // 0,8,16,24 staging col

  int gm = m0 + lr; if (gm > M - 1) gm = M - 1;           // clamp (stores guarded)
  const unsigned short* ga = A + (size_t)gm * K + lc;
  const unsigned short* gb = W + (size_t)(n0 + lr) * K + lc;
  const int nsteps = K >> 5;

#ifdef HAVE_ASYNC_LDS
  // ---- double-buffered async global->LDS pipeline (ASYNCcnt-tracked) ----
  __builtin_amdgcn_global_load_async_to_lds_b128(AS1CAST(ga), AS3CAST(&As[0][lr][lc]), 0, 0);
  __builtin_amdgcn_global_load_async_to_lds_b128(AS1CAST(gb), AS3CAST(&Bs[0][lr][lc]), 0, 0);
  for (int s = 0; s < nsteps; ++s) {
    const int cur = s & 1;
    if (s + 1 < nsteps) {
      const int nxt = cur ^ 1;
      __builtin_amdgcn_global_load_async_to_lds_b128(
          AS1CAST(ga + (size_t)(s + 1) * 32), AS3CAST(&As[nxt][lr][lc]), 0, 0);
      __builtin_amdgcn_global_load_async_to_lds_b128(
          AS1CAST(gb + (size_t)(s + 1) * 32), AS3CAST(&Bs[nxt][lr][lc]), 0, 0);
      __builtin_amdgcn_s_wait_asynccnt(2);  // in-order: stage s complete
    } else {
      __builtin_amdgcn_s_wait_asynccnt(0);
    }
    __syncthreads();                        // stage-s tiles visible to all waves
    wmma_tiles(As[cur], Bs[cur], lane, wm, wn, acc);
    __syncthreads();                        // all reads done before buf reuse
  }
#else
  // ---- fallback: synchronous staging, single buffer ----
  for (int s = 0; s < nsteps; ++s) {
    *(uint4*)(&As[0][lr][lc]) = *(const uint4*)(ga + (size_t)s * 32);
    *(uint4*)(&Bs[0][lr][lc]) = *(const uint4*)(gb + (size_t)s * 32);
    __syncthreads();
    wmma_tiles(As[0], Bs[0], lane, wm, wn, acc);
    __syncthreads();
  }
#endif

  #pragma unroll
  for (int t = 0; t < 2; ++t) {
    #pragma unroll
    for (int r = 0; r < 8; ++r) {
      const int m = m0 + wm * 16 + r + ((lane >= 16) ? 8 : 0);
      const int n = n0 + wn * 32 + t * 16 + (lane & 15);
      if (m < M) {
        const float val = acc[t].f[r] + bias[n];
        if (mode == 0) {
          const int b = m / NTOK, ntok = m % NTOK;
          const int sel = n >> 9, rem = n & 511;
          const int head = rem >> 6, d = rem & 63;
          const size_t idx = ((size_t)(b * NH + head) * NTOK + ntok) * CH + d;
          const unsigned short bv = f2bf(val);
          if (sel == 0) qb[idx] = bv;
          else if (sel == 1) kbuf[idx] = bv;
          else vbuf[idx] = bv;
        } else {
          outf[(size_t)m * Nn + n] = val;
        }
      }
    }
  }
}

// ---------------------------------------------------------------------------
// Per-column (b,h,d) softmax stats over N tokens: max and sum(exp(x-max))
// ---------------------------------------------------------------------------
__global__ void k_softmax_stats(const unsigned short* __restrict__ kbuf,
                                float* __restrict__ colM, float* __restrict__ colS){
  __shared__ float red[BDIM];
  const int col = blockIdx.x;                 // bh*64 + d
  const int bh = col >> 6, d = col & 63;
  const unsigned short* base = kbuf + (size_t)bh * NTOK * CH + d;
  const int tid = threadIdx.x;

  float mx = -1e30f;
  for (int n = tid; n < NTOK; n += BDIM) mx = fmaxf(mx, bf2f(base[(size_t)n * CH]));
  red[tid] = mx; __syncthreads();
  for (int s = BDIM >> 1; s > 0; s >>= 1) {
    if (tid < s) red[tid] = fmaxf(red[tid], red[tid + s]);
    __syncthreads();
  }
  mx = red[0]; __syncthreads();

  float sm = 0.0f;
  for (int n = tid; n < NTOK; n += BDIM) sm += __expf(bf2f(base[(size_t)n * CH]) - mx);
  red[tid] = sm; __syncthreads();
  for (int s = BDIM >> 1; s > 0; s >>= 1) {
    if (tid < s) red[tid] += red[tid + s];
    __syncthreads();
  }
  if (tid == 0) { colM[col] = mx; colS[col] = red[0]; }
}

// ---------------------------------------------------------------------------
// kv[bh] (64x64) = softmax(k)^T @ v, softmax fused: A = exp(k-max) built in
// the loader, row-scaled by 1/sum in the epilogue. One block per (b,h).
// ---------------------------------------------------------------------------
__global__ void gemm_kv(const unsigned short* __restrict__ kbuf,
                        const unsigned short* __restrict__ vbuf,
                        const float* __restrict__ colM,
                        const float* __restrict__ colS,
                        unsigned short* __restrict__ kvb){
  const int bh = blockIdx.x;
  const int tid = threadIdx.x, lane = tid & 31, w = tid >> 5;
  const int tm = w >> 1;          // 0..3  kdim tile
  const int th = w & 1;           // 0..1  d tile pair
  const unsigned short* kbase = kbuf + (size_t)bh * NTOK * CH;
  const unsigned short* vbase = vbuf + (size_t)bh * NTOK * CH;

  Acc acc[2];
  #pragma unroll
  for (int t = 0; t < 2; ++t)
    #pragma unroll
    for (int i = 0; i < 8; ++i) acc[t].f[i] = 0.0f;

  const int kd = tm * 16 + (lane & 15);       // A-fragment row (kdim)
  const float sub = colM[bh * 64 + kd];

  for (int nk = 0; nk < NTOK; nk += 32) {
    Frag af;
    const int k0 = (lane < 16) ? 0 : 8;
    #pragma unroll
    for (int j = 0; j < 8; ++j) {
      const int n1 = nk + k0 + j;
      const int n2 = n1 + 16;
      af.u[j]     = (n1 < NTOK) ? f2bf(__expf(bf2f(kbase[(size_t)n1 * CH + kd]) - sub)) : 0;
      af.u[j + 8] = (n2 < NTOK) ? f2bf(__expf(bf2f(kbase[(size_t)n2 * CH + kd]) - sub)) : 0;
    }
    #pragma unroll
    for (int t = 0; t < 2; ++t) {
      Frag bfr;
      const int dl = th * 32 + t * 16 + (lane & 15);
      const int nb = nk + ((lane < 16) ? 0 : 8);
      #pragma unroll
      for (int j = 0; j < 16; ++j) {
        const int n = nb + j + ((j >= 8) ? 8 : 0);
        bfr.u[j] = (n < NTOK) ? vbase[(size_t)n * CH + dl] : 0;
      }
      acc[t].v = __builtin_amdgcn_wmma_f32_16x16x32_bf16(
          false, af.v, false, bfr.v, (short)0, acc[t].v, false, false);
    }
  }

  #pragma unroll
  for (int t = 0; t < 2; ++t) {
    #pragma unroll
    for (int r = 0; r < 8; ++r) {
      const int kdo = tm * 16 + r + ((lane >= 16) ? 8 : 0);
      const int dl  = th * 32 + t * 16 + (lane & 15);
      const float inv = 1.0f / colS[bh * 64 + kdo];
      kvb[((size_t)bh * 64 + kdo) * 64 + dl] = f2bf(acc[t].f[r] * inv);
    }
  }
}

// ---------------------------------------------------------------------------
// Depthwise conv on v image (3x3 / 5x5 / 7x7 channel groups), reading v in
// (B,h,N,Ch) bf16 layout (fits entirely in 192MB L2), f32 output (B,C,H,W).
// ---------------------------------------------------------------------------
__global__ void dwconv(const unsigned short* __restrict__ vbuf,
                       const float* __restrict__ w3, const float* __restrict__ b3,
                       const float* __restrict__ w5, const float* __restrict__ b5,
                       const float* __restrict__ w7, const float* __restrict__ b7,
                       float* __restrict__ convv){
  const long long idx = (long long)blockIdx.x * blockDim.x + threadIdx.x;
  const long long total = (long long)BB * CC * HWP;
  if (idx >= total) return;
  const int p = (int)(idx % HWP);
  const int c = (int)((idx / HWP) % CC);
  const int b = (int)(idx / ((long long)HWP * CC));
  const int y = p / 56, x = p % 56;

  int ks; const float* wp; float acc;
  if (c < 128)      { ks = 3; wp = w3 + c * 9;          acc = b3[c]; }
  else if (c < 320) { ks = 5; wp = w5 + (c - 128) * 25; acc = b5[c - 128]; }
  else              { ks = 7; wp = w7 + (c - 320) * 49; acc = b7[c - 320]; }
  const int pad = ks >> 1;
  const int head = c >> 6, d = c & 63;
  const unsigned short* vbase = vbuf + ((size_t)(b * NH + head) * NTOK + 1) * CH + d;

  for (int kh = 0; kh < ks; ++kh) {
    const int yy = y + kh - pad;
    if (yy < 0 || yy >= 56) continue;
    for (int kw = 0; kw < ks; ++kw) {
      const int xx = x + kw - pad;
      if (xx < 0 || xx >= 56) continue;
      acc += wp[kh * ks + kw] * bf2f(vbase[(size_t)(yy * 56 + xx) * CH]);
    }
  }
  convv[idx] = acc;
}

// ---------------------------------------------------------------------------
// factor_att = q @ kv per (b,h), fused with CRPE combine:
//   out = 0.125*factor_att + (tok>0 ? q*conv_v : 0)  -> bf16 proj input
// ---------------------------------------------------------------------------
__global__ void attn_combine(const unsigned short* __restrict__ qb,
                             const unsigned short* __restrict__ kvb,
                             const float* __restrict__ convv,
                             unsigned short* __restrict__ x2){
  const int bh = blockIdx.y;
  const int b = bh >> 3, head = bh & 7;
  const int tid = threadIdx.x, lane = tid & 31, w = tid >> 5;
  const int mt = blockIdx.x * 2 + (w >> 2);     // token tile (wave-uniform)
  const int nt = w & 3;                         // d tile
  if (mt >= MT197) return;                      // uniform per wave -> EXEC stays full

  const unsigned short* qbase = qb + (size_t)bh * NTOK * CH;

  Acc acc;
  #pragma unroll
  for (int i = 0; i < 8; ++i) acc.f[i] = 0.0f;

  #pragma unroll
  for (int kk = 0; kk < 64; kk += 32) {
    Frag af;
    int tok = mt * 16 + (lane & 15);
    if (tok > NTOK - 1) tok = NTOK - 1;         // clamp; OOB rows never stored
    const int k0 = kk + ((lane < 16) ? 0 : 8);
    unpack8(*(const uint4*)(qbase + (size_t)tok * CH + k0),      &af.u[0]);
    unpack8(*(const uint4*)(qbase + (size_t)tok * CH + k0 + 16), &af.u[8]);

    Frag bfr;
    const int dl = nt * 16 + (lane & 15);
    const int kb0 = kk + ((lane < 16) ? 0 : 8);
    #pragma unroll
    for (int j = 0; j < 16; ++j)
      bfr.u[j] = kvb[((size_t)bh * 64 + kb0 + j + ((j >= 8) ? 8 : 0)) * 64 + dl];

    acc.v = __builtin_amdgcn_wmma_f32_16x16x32_bf16(
        false, af.v, false, bfr.v, (short)0, acc.v, false, false);
  }

  const float scale = 0.125f;                   // 64^-0.5
  #pragma unroll
  for (int r = 0; r < 8; ++r) {
    const int tok = mt * 16 + r + ((lane >= 16) ? 8 : 0);
    if (tok < NTOK) {
      const int dg = nt * 16 + (lane & 15);
      float val = scale * acc.f[r];
      if (tok > 0) {
        const float qv = bf2f(qbase[(size_t)tok * CH + dg]);
        const float cv = convv[((size_t)(b * CC) + head * CH + dg) * HWP + (tok - 1)];
        val += qv * cv;
      }
      x2[((size_t)(b * NTOK + tok)) * CC + head * CH + dg] = f2bf(val);
    }
  }
}

// ---------------------------------------------------------------------------
extern "C" void kernel_launch(void* const* d_in, const int* in_sizes, int n_in,
                              void* d_out, int out_size, void* d_ws, size_t ws_size,
                              hipStream_t stream){
  (void)in_sizes; (void)n_in; (void)out_size; (void)ws_size;
  const float* x      = (const float*)d_in[0];
  const float* qkv_w  = (const float*)d_in[1];
  const float* qkv_b  = (const float*)d_in[2];
  const float* proj_w = (const float*)d_in[3];
  const float* proj_b = (const float*)d_in[4];
  const float* w3 = (const float*)d_in[5];
  const float* b3 = (const float*)d_in[6];
  const float* w5 = (const float*)d_in[7];
  const float* b5 = (const float*)d_in[8];
  const float* w7 = (const float*)d_in[9];
  const float* b7 = (const float*)d_in[10];
  float* out = (float*)d_out;

  char* base = (char*)d_ws;
  size_t off = 0;
  auto take = [&](size_t bytes) -> char* {
    char* p = base + off;
    off += (bytes + 255) & ~(size_t)255;
    return p;
  };
  unsigned short* xb     = (unsigned short*)take((size_t)MALL * CC * 2);
  unsigned short* wqkvb  = (unsigned short*)take((size_t)3 * CC * CC * 2);
  unsigned short* wprojb = (unsigned short*)take((size_t)CC * CC * 2);
  unsigned short* qb     = (unsigned short*)take((size_t)MALL * CC * 2);
  unsigned short* kb     = (unsigned short*)take((size_t)MALL * CC * 2);
  unsigned short* vb     = (unsigned short*)take((size_t)MALL * CC * 2);
  float*          colM   = (float*)take((size_t)BB * NH * CH * 4);
  float*          colS   = (float*)take((size_t)BB * NH * CH * 4);
  unsigned short* kvb    = (unsigned short*)take((size_t)BB * NH * CH * CH * 2);
  float*          convv  = (float*)take((size_t)BB * CC * HWP * 4);
  unsigned short* x2     = (unsigned short*)take((size_t)MALL * CC * 2);

  const int thr = BDIM;

  // 1. f32 -> bf16 conversions (8 elems/thread)
  cvt_f32_bf16<<<2048, thr, 0, stream>>>(x, xb, (long long)MALL * CC / 8);
  cvt_f32_bf16<<<256,  thr, 0, stream>>>(qkv_w, wqkvb, (long long)3 * CC * CC / 8);
  cvt_f32_bf16<<<128,  thr, 0, stream>>>(proj_w, wprojb, (long long)CC * CC / 8);

  // 2. qkv GEMM (M=50192, N=1536, K=512), scatter into q/k/v
  dim3 g1(3 * CC / 64, (MALL + 63) / 64);
  gemm_bf16_tn<<<g1, thr, 0, stream>>>(xb, wqkvb, qkv_b, MALL, 3 * CC, CC, 0,
                                       qb, kb, vb, nullptr);

  // 3. softmax stats over tokens per (b,h,d)
  k_softmax_stats<<<BB * NH * CH, thr, 0, stream>>>(kb, colM, colS);

  // 4. kv = softmax(k)^T @ v, 128 small WMMA GEMMs (softmax fused)
  gemm_kv<<<BB * NH, thr, 0, stream>>>(kb, vb, colM, colS, kvb);

  // 5. depthwise conv rel-pos on v image
  const long long ctotal = (long long)BB * CC * HWP;
  dwconv<<<(unsigned)((ctotal + thr - 1) / thr), thr, 0, stream>>>(
      vb, w3, b3, w5, b5, w7, b7, convv);

  // 6. factor_att GEMM + CRPE combine -> bf16 proj input
  dim3 g2((MT197 + 1) / 2, BB * NH);
  attn_combine<<<g2, thr, 0, stream>>>(qb, kvb, convv, x2);

  // 7. proj GEMM -> f32 output
  dim3 g3(CC / 64, (MALL + 63) / 64);
  gemm_bf16_tn<<<g3, thr, 0, stream>>>(x2, wprojb, proj_b, MALL, CC, CC, 1,
                                       nullptr, nullptr, nullptr, out);
}
